// GeoTransformer_4604204941601
// MI455X (gfx1250) — matrix operand order repaired and checked by reading, not compiled, not run on hardware
//
#include <hip/hip_runtime.h>
#include <math.h>

// ---------------- problem constants ----------------
constexpr int Bb   = 4;
constexpr int Nn   = 1024;
constexpr int Dd   = 512;
constexpr int Hh   = 8;
constexpr int DKk  = 64;
constexpr int DFFf = 2048;
constexpr int HIDh = 50;
constexpr int HIDP = 64;   // padded hidden
constexpr int Ll   = 3;
constexpr int BN   = Bb * Nn;
constexpr float NEGV = -100000000.0f;

typedef __attribute__((ext_vector_type(16))) _Float16 v16h;
typedef __attribute__((ext_vector_type(8)))  _Float16 v8h;
typedef __attribute__((ext_vector_type(8)))  float    v8f;
typedef __attribute__((ext_vector_type(4)))  float    v4f;

#define DEV static __device__ __forceinline__

// ---------------- WMMA wrapper ----------------
DEV v8f wmma32(v16h a, v16h b, v8f c) {
  return __builtin_amdgcn_wmma_f32_16x16x32_f16(
      false, a, false, b, (short)0, c, false, false);
}

// ---------------- operand loaders (wave32; A and B share contiguous k order k = 16h + e) ----------------
// A 16x32 f16: lanes 0-15 row M=lane, K = klo+{0..15} with klo=0;
// lanes 16-31 row M=lane-16, klo=16. Vector element e -> K=klo+e (contiguous).

DEV v16h load_a_f32(const float* A, int lda, int k0, int lane) {
  int m   = lane & 15;
  int klo = (lane < 16) ? 0 : 16;
  const float* ar = A + (size_t)m * lda + k0 + klo;
  v4f x0 = *(const v4f*)(ar + 0);
  v4f x1 = *(const v4f*)(ar + 4);
  v4f x2 = *(const v4f*)(ar + 8);
  v4f x3 = *(const v4f*)(ar + 12);
  v16h r;
#pragma unroll
  for (int i = 0; i < 4; ++i) {
    r[i]      = (_Float16)x0[i];
    r[4 + i]  = (_Float16)x1[i];
    r[8 + i]  = (_Float16)x2[i];
    r[12 + i] = (_Float16)x3[i];
  }
  return r;
}

DEV v16h load_a_h(const _Float16* A, int lda, int k0, int lane) {
  int m   = lane & 15;
  int klo = (lane < 16) ? 0 : 16;
  const _Float16* ar = A + (size_t)m * lda + k0 + klo;
  v8h lo = *(const v8h*)(ar);
  v8h hi = *(const v8h*)(ar + 8);
  v16h r;
#pragma unroll
  for (int i = 0; i < 8; ++i) { r[i] = lo[i]; r[8 + i] = hi[i]; }
  return r;
}

// B 32x16 f16: lane column n = lane&15 ; lanes<16 K=k0+0..15, lanes>=16 K=k0+16..31
// (vector element e -> K = kb+e, contiguous 16 values).
DEV v16h load_b_w(const float* W, int ldw, int n0, int k0, int lane) {
  int n  = n0 + (lane & 15);
  int kb = k0 + ((lane < 16) ? 0 : 16);
  const float* wr = W + (size_t)n * ldw + kb;
  v4f x0 = *(const v4f*)(wr + 0);
  v4f x1 = *(const v4f*)(wr + 4);
  v4f x2 = *(const v4f*)(wr + 8);
  v4f x3 = *(const v4f*)(wr + 12);
  v16h r;
#pragma unroll
  for (int i = 0; i < 4; ++i) {
    r[i]      = (_Float16)x0[i];
    r[4 + i]  = (_Float16)x1[i];
    r[8 + i]  = (_Float16)x2[i];
    r[12 + i] = (_Float16)x3[i];
  }
  return r;
}

// B operand, K dim contiguous within a row-major f16 row (scores: column n = token j)
DEV v16h load_b_h_rows(const _Float16* Kmat, int lda, int j0, int d0, int lane) {
  int j  = j0 + (lane & 15);
  int dd = d0 + ((lane < 16) ? 0 : 16);
  return *(const v16h*)(Kmat + (size_t)j * lda + dd);
}

// B operand from transposed V buffer vT[d][j] (K dim = j contiguous)
DEV v16h load_b_h_cols(const _Float16* Vt, int ldv, int j0, int lane) {
  int dcol = lane & 15;
  int jb   = j0 + ((lane < 16) ? 0 : 16);
  return *(const v16h*)(Vt + (size_t)dcol * ldv + jb);
}

DEV float geluf(float x) { return 0.5f * x * (1.f + erff(x * 0.70710678118654752f)); }

DEV float redmax16(float v) {
#pragma unroll
  for (int m = 1; m < 16; m <<= 1) v = fmaxf(v, __shfl_xor(v, m, 16));
  return v;
}
DEV float redadd16(float v) {
#pragma unroll
  for (int m = 1; m < 16; m <<= 1) v += __shfl_xor(v, m, 16);
  return v;
}

// ================= 0. pad dist-MLP weights 50 -> 64 (one shot) =================
__global__ void gt_prep(const float* __restrict__ dw0, const float* __restrict__ db0,
                        const float* __restrict__ dw1, const float* __restrict__ db1,
                        const float* __restrict__ dw2, const float* __restrict__ db2,
                        const float* __restrict__ dw3,
                        float* __restrict__ w0p, float* __restrict__ b0p,
                        float* __restrict__ w1p, float* __restrict__ b1p,
                        float* __restrict__ w2p, float* __restrict__ b2p,
                        float* __restrict__ w3p) {
  int tid = blockIdx.x * blockDim.x + threadIdx.x;
  int stride = gridDim.x * blockDim.x;
  for (int idx = tid; idx < Ll * HIDP * HIDP; idx += stride) {
    int l = idx / (HIDP * HIDP), rem = idx % (HIDP * HIDP);
    int n = rem / HIDP, k = rem % HIDP;
    bool in = (n < HIDh) && (k < HIDh);
    float v1 = in ? dw1[((size_t)l * HIDh + n) * HIDh + k] : 0.f;
    float v2 = in ? dw2[((size_t)l * HIDh + n) * HIDh + k] : 0.f;
    w1p[idx] = v1;
    w2p[idx] = v2;
  }
  for (int idx = tid; idx < Ll * HIDP; idx += stride) {
    int l = idx / HIDP, k = idx % HIDP;
    bool in = (k < HIDh);
    w0p[idx] = in ? dw0[l * HIDh + k] : 0.f;
    b0p[idx] = in ? db0[l * HIDh + k] : 0.f;
    b1p[idx] = in ? db1[l * HIDh + k] : 0.f;
    b2p[idx] = in ? db2[l * HIDh + k] : 0.f;
    w3p[idx] = in ? dw3[l * HIDh + k] : 0.f;
  }
}

// ================= 1. pairwise inverse distances =================
__global__ void gt_dists(const float* __restrict__ R, const int* __restrict__ mask,
                         float* __restrict__ dists) {
  int bi = blockIdx.x;  // b*N + i
  int b  = bi >> 10;
  int i  = bi & (Nn - 1);
  float xi = R[(size_t)bi * 3 + 0];
  float yi = R[(size_t)bi * 3 + 1];
  float zi = R[(size_t)bi * 3 + 2];
  int mi = mask[bi];
  for (int j = threadIdx.x; j < Nn; j += blockDim.x) {
    size_t rj = (size_t)(b * Nn + j) * 3;
    float dx = xi - R[rj + 0];
    float dy = yi - R[rj + 1];
    float dz = zi - R[rj + 2];
    float d2 = dx * dx + dy * dy + dz * dz;
    bool valid = (i != j) && (mi > 0) && (mask[b * Nn + j] > 0);
    dists[(size_t)bi * Nn + j] = valid ? (1.0f / sqrtf(d2)) : 0.0f;
  }
}

// ================= 2. embedding + positional encoding =================
__global__ void gt_embed(const int* __restrict__ charges, const float* __restrict__ dists,
                         const float* __restrict__ pw0, const float* __restrict__ pb0,
                         const float* __restrict__ pw1, const float* __restrict__ pb1,
                         const float* __restrict__ pew, const float* __restrict__ peb,
                         const float* __restrict__ aemb, float* __restrict__ x) {
  __shared__ float red[256];
  int bi  = blockIdx.x;
  int b   = bi >> 10;
  int tid = threadIdx.x;
  float cmi = (charges[bi] > 0) ? 1.f : 0.f;
  float partial = 0.f;
  for (int j = tid; j < Nn; j += 256) {
    float d   = dists[(size_t)bi * Nn + j];
    float cmj = (charges[b * Nn + j] > 0) ? 1.f : 0.f;
    float accu = pb1[0];
#pragma unroll 10
    for (int k = 0; k < HIDh; ++k)
      accu += geluf(d * pw0[k] + pb0[k]) * pw1[k];
    partial += geluf(accu) * cmi * cmj;
  }
  red[tid] = partial;
  __syncthreads();
  for (int st = 128; st > 0; st >>= 1) {
    if (tid < st) red[tid] += red[tid + st];
    __syncthreads();
  }
  float s  = red[0];
  int   ch = charges[bi];
  for (int d = tid; d < Dd; d += 256)
    x[(size_t)bi * Dd + d] = aemb[(size_t)ch * Dd + d] + s * pew[d] + peb[d];
}

// ================= 3. layernorm (one wave per row of 512) =================
__global__ void gt_ln(const float* __restrict__ X, const float* __restrict__ g,
                      const float* __restrict__ bt, float* __restrict__ Y) {
  int lane = threadIdx.x;
  int row  = blockIdx.x * 8 + threadIdx.y;
  const float* xr = X + (size_t)row * Dd;
  float v[16];
  float s = 0.f;
#pragma unroll
  for (int t = 0; t < 16; ++t) { v[t] = xr[lane + 32 * t]; s += v[t]; }
#pragma unroll
  for (int m = 1; m < 32; m <<= 1) s += __shfl_xor(s, m, 32);
  float mean = s * (1.f / Dd);
  float q = 0.f;
#pragma unroll
  for (int t = 0; t < 16; ++t) { float d = v[t] - mean; q += d * d; }
#pragma unroll
  for (int m = 1; m < 32; m <<= 1) q += __shfl_xor(q, m, 32);
  float inv = rsqrtf(q * (1.f / Dd) + 1e-5f);
#pragma unroll
  for (int t = 0; t < 16; ++t) {
    int c = lane + 32 * t;
    Y[(size_t)row * Dd + c] = (v[t] - mean) * inv * g[c] + bt[c];
  }
}

// ================= 4. fused QKV projection (shared A tile, 3 WMMA/K-step) ======
__global__ void gt_qkv(const float* __restrict__ A, const float* __restrict__ Wq,
                       const float* __restrict__ Wk, const float* __restrict__ Wv,
                       _Float16* __restrict__ qo, _Float16* __restrict__ ko,
                       _Float16* __restrict__ vo) {
  int lane = threadIdx.x;
  int tile = blockIdx.x * 8 + threadIdx.y;
  const int NTN = Dd / 16;
  int mt = tile / NTN, nt = tile % NTN;
  const float* Ar = A + (size_t)mt * 16 * Dd;
  v8f cq = {}, ck = {}, cv = {};
  for (int k0 = 0; k0 < Dd; k0 += 32) {
    v16h a = load_a_f32(Ar, Dd, k0, lane);
    cq = wmma32(a, load_b_w(Wq, Dd, nt * 16, k0, lane), cq);
    ck = wmma32(a, load_b_w(Wk, Dd, nt * 16, k0, lane), ck);
    cv = wmma32(a, load_b_w(Wv, Dd, nt * 16, k0, lane), cv);
  }
  int col = lane & 15, rbase = (lane < 16) ? 0 : 8;
  int n = nt * 16 + col;
  int hh = n >> 6, dd = n & 63;
#pragma unroll
  for (int r = 0; r < 8; ++r) {
    int row = mt * 16 + rbase + r;
    qo[(size_t)row * Dd + n] = (_Float16)cq[r];
    ko[(size_t)row * Dd + n] = (_Float16)ck[r];
    int b = row >> 10, i = row & (Nn - 1);
    vo[((size_t)(b * Hh + hh) * DKk + dd) * Nn + i] = (_Float16)cv[r];
  }
}

// ================= 5. fused distance MLP (padded 64-wide weights) =================
__global__ void gt_distmlp(const float* __restrict__ dists, const int* __restrict__ mask,
                           const float* __restrict__ w0p, const float* __restrict__ b0p,
                           const float* __restrict__ w1p, const float* __restrict__ b1p,
                           const float* __restrict__ w2p, const float* __restrict__ b2p,
                           const float* __restrict__ w3p, const float* __restrict__ db3,
                           float* __restrict__ sdv) {
  __shared__ _Float16 sm[8][16][HIDP];
  int lane = threadIdx.x, w = threadIdx.y;
  size_t p0 = ((size_t)blockIdx.x * 8 + w) * 16;
  int m   = lane & 15;
  int klo = (lane < 16) ? 0 : 16;
  float dm = dists[p0 + m];

  // layer0: relu(d*w0+b0) built directly in A layout (padded cols give relu(0)=0)
  v16h a0, a1;
#pragma unroll
  for (int v = 0; v < 8; ++v) {
    int off = 2 * v;
    int kA = klo + off;        // 0..31
    int kB = 32 + klo + off;   // 32..63 (padded)
    a0[2 * v]     = (_Float16)fmaxf(dm * w0p[kA]     + b0p[kA],     0.f);
    a0[2 * v + 1] = (_Float16)fmaxf(dm * w0p[kA + 1] + b0p[kA + 1], 0.f);
    a1[2 * v]     = (_Float16)fmaxf(dm * w0p[kB]     + b0p[kB],     0.f);
    a1[2 * v + 1] = (_Float16)fmaxf(dm * w0p[kB + 1] + b0p[kB + 1], 0.f);
  }

  _Float16* Tp = &sm[w][0][0];
  int col = lane & 15, rbase = (lane < 16) ? 0 : 8;
#pragma unroll
  for (int layer = 0; layer < 2; ++layer) {
    const float* Wd = layer ? w2p : w1p;
    const float* Bd = layer ? b2p : b1p;
#pragma unroll
    for (int nc = 0; nc < 4; ++nc) {
      v8f c = {};
      c = wmma32(a0, load_b_w(Wd, HIDP, nc * 16, 0,  lane), c);
      c = wmma32(a1, load_b_w(Wd, HIDP, nc * 16, 32, lane), c);
      int n = nc * 16 + col;
      float bb = Bd[n];
#pragma unroll
      for (int r = 0; r < 8; ++r)
        Tp[(rbase + r) * HIDP + n] = (_Float16)fmaxf(c[r] + bb, 0.f);
    }
    a0 = load_a_h(Tp, HIDP, 0, lane);
    a1 = load_a_h(Tp, HIDP, 32, lane);
  }

  // layer3: dot with padded w3 + mask
  if (lane < 16) {
    float acc = db3[0];
#pragma unroll
    for (int k = 0; k < HIDP; ++k) acc += (float)Tp[lane * HIDP + k] * w3p[k];
    size_t p   = p0 + lane;
    size_t rem = p & ((size_t)Nn * Nn - 1);
    int b = (int)(p >> 20);
    int i = (int)(rem >> 10);
    int j = (int)(rem & (Nn - 1));
    bool valid = (mask[b * Nn + i] > 0) && (mask[b * Nn + j] > 0);
    sdv[p] = valid ? acc : 0.f;
  }
}

// ================= 6. flash attention with sd^2 weighting =================
__global__ void gt_attn(const _Float16* __restrict__ qm, const _Float16* __restrict__ km,
                        const _Float16* __restrict__ vtm, const float* __restrict__ sdv,
                        const int* __restrict__ mask, float* __restrict__ ob) {
  __shared__ _Float16 pb[8][16][32];
  const int lane = threadIdx.x, w = threadIdx.y;
  int gw = blockIdx.x * 8 + w;
  const int NT = Nn / 16;
  int it = gw % NT, bh = gw / NT;
  int b = bh / Hh, h = bh % Hh;
  int i0 = it * 16;
  int col = lane & 15, rbase = (lane < 16) ? 0 : 8;

  const _Float16* qbase = qm + ((size_t)(b * Nn + i0) * Dd + h * DKk);
  v16h qa0 = load_a_h(qbase, Dd, 0, lane);
  v16h qa1 = load_a_h(qbase, Dd, 32, lane);
  const _Float16* kbase = km + ((size_t)(b * Nn) * Dd + h * DKk);
  const _Float16* vbase = vtm + ((size_t)(b * Hh + h) * DKk) * (size_t)Nn;

  float rowm[8], rowl[8], rmk[8];
  v8f acc[4] = {};
#pragma unroll
  for (int r = 0; r < 8; ++r) {
    rowm[r] = -3.0e38f;
    rowl[r] = 0.f;
    rmk[r]  = (mask[b * Nn + i0 + rbase + r] > 0) ? 1.f : 0.f;
  }
  const float scale = 1.f / (8.f + 1e-8f);  // 1/(sqrt(DK)+1e-8)

  for (int jt = 0; jt < Nn / 32; ++jt) {
    int j0 = jt * 32;
    v8f s0 = {}, s1 = {};
    s0 = wmma32(qa0, load_b_h_rows(kbase, Dd, j0,      0,  lane), s0);
    s0 = wmma32(qa1, load_b_h_rows(kbase, Dd, j0,      32, lane), s0);
    s1 = wmma32(qa0, load_b_h_rows(kbase, Dd, j0 + 16, 0,  lane), s1);
    s1 = wmma32(qa1, load_b_h_rows(kbase, Dd, j0 + 16, 32, lane), s1);

    int jc0 = j0 + col, jc1 = j0 + 16 + col;
    float cm0 = (mask[b * Nn + jc0] > 0) ? 1.f : 0.f;
    float cm1 = (mask[b * Nn + jc1] > 0) ? 1.f : 0.f;

#pragma unroll
    for (int r = 0; r < 8; ++r) {
      int i = i0 + rbase + r;
      float v0 = s0[r] * scale;
      float v1 = s1[r] * scale;
      if (cm0 == 0.f || rmk[r] == 0.f || i == jc0) v0 = NEGV;
      if (cm1 == 0.f || rmk[r] == 0.f || i == jc1) v1 = NEGV;
      float mx   = redmax16(fmaxf(v0, v1));
      float newm = fmaxf(rowm[r], mx);
      float corr = __expf(rowm[r] - newm);
      float p0 = __expf(v0 - newm);
      float p1 = __expf(v1 - newm);
      float ps = redadd16(p0 + p1);
      rowl[r] = rowl[r] * corr + ps;
      rowm[r] = newm;
#pragma unroll
      for (int c = 0; c < 4; ++c) acc[c][r] *= corr;
      float d0 = sdv[(size_t)(b * Nn + i) * Nn + jc0];
      float d1 = sdv[(size_t)(b * Nn + i) * Nn + jc1];
      pb[w][rbase + r][col]      = (_Float16)(p0 * d0 * d0);
      pb[w][rbase + r][16 + col] = (_Float16)(p1 * d1 * d1);
    }
    v16h pa = load_a_h(&pb[w][0][0], 32, 0, lane);
#pragma unroll
    for (int c = 0; c < 4; ++c)
      acc[c] = wmma32(pa, load_b_h_cols(vbase + (size_t)(c * 16) * Nn, Nn, j0, lane), acc[c]);
  }

#pragma unroll
  for (int r = 0; r < 8; ++r) {
    float inv = (rowl[r] > 0.f) ? (rmk[r] / rowl[r]) : 0.f;
    size_t orow = (size_t)(b * Nn + i0 + rbase + r) * Dd + h * DKk;
#pragma unroll
    for (int c = 0; c < 4; ++c) ob[orow + c * 16 + col] = acc[c][r] * inv;
  }
}

// ====== 7. strip GEMM: 16x64 output strip per wave, optional residual ======
__global__ void gt_gemm(const float* __restrict__ A, const float* __restrict__ W,
                        const float* __restrict__ Res, float* __restrict__ Out,
                        int Nout, int K) {
  int lane  = threadIdx.x;
  int strip = blockIdx.x * 8 + threadIdx.y;
  int nsn = Nout / 64;                   // strips per row
  int mt = strip / nsn, ns = strip % nsn;
  int nt0 = ns * 4;
  const float* Ar = A + (size_t)mt * 16 * K;
  v8f acc[4] = {};
  for (int k0 = 0; k0 < K; k0 += 32) {
    v16h a = load_a_f32(Ar, K, k0, lane);
#pragma unroll
    for (int c = 0; c < 4; ++c)
      acc[c] = wmma32(a, load_b_w(W, K, (nt0 + c) * 16, k0, lane), acc[c]);
  }
  int col = lane & 15, rbase = (lane < 16) ? 0 : 8;
#pragma unroll
  for (int c = 0; c < 4; ++c) {
    int n = (nt0 + c) * 16 + col;
#pragma unroll
    for (int r = 0; r < 8; ++r) {
      size_t idx = (size_t)(mt * 16 + rbase + r) * Nout + n;
      float v = acc[c][r];
      if (Res) v += Res[idx];
      Out[idx] = v;
    }
  }
}

// ====== 8. gated FFN strip: u = gelu(h@g1^T) * (h@g2^T), 16x32 per wave ======
__global__ void gt_gate(const float* __restrict__ A, const float* __restrict__ W1,
                        const float* __restrict__ W2, float* __restrict__ U) {
  int lane  = threadIdx.x;
  int strip = blockIdx.x * 8 + threadIdx.y;
  const int NSN = DFFf / 32;             // 64 strips (2 n-tiles each)
  int mt = strip / NSN, ns = strip % NSN;
  int nt0 = ns * 2;
  const float* Ar = A + (size_t)mt * 16 * Dd;
  v8f c1[2] = {}, c2[2] = {};
  for (int k0 = 0; k0 < Dd; k0 += 32) {
    v16h a = load_a_f32(Ar, Dd, k0, lane);
#pragma unroll
    for (int c = 0; c < 2; ++c) {
      c1[c] = wmma32(a, load_b_w(W1, Dd, (nt0 + c) * 16, k0, lane), c1[c]);
      c2[c] = wmma32(a, load_b_w(W2, Dd, (nt0 + c) * 16, k0, lane), c2[c]);
    }
  }
  int col = lane & 15, rbase = (lane < 16) ? 0 : 8;
#pragma unroll
  for (int c = 0; c < 2; ++c) {
    int n = (nt0 + c) * 16 + col;
#pragma unroll
    for (int r = 0; r < 8; ++r)
      U[(size_t)(mt * 16 + rbase + r) * DFFf + n] = geluf(c1[c][r]) * c2[c][r];
  }
}

// ================= host-side orchestration =================
extern "C" void kernel_launch(void* const* d_in, const int* in_sizes, int n_in,
                              void* d_out, int out_size, void* d_ws, size_t ws_size,
                              hipStream_t stream) {
  (void)in_sizes; (void)n_in; (void)out_size; (void)ws_size;
  const int*   charges = (const int*)d_in[0];
  const int*   amask   = (const int*)d_in[1];
  const float* R       = (const float*)d_in[2];
  const float* aemb    = (const float*)d_in[3];
  const float* pw0     = (const float*)d_in[4];
  const float* pb0     = (const float*)d_in[5];
  const float* pw1     = (const float*)d_in[6];
  const float* pb1     = (const float*)d_in[7];
  const float* pew     = (const float*)d_in[8];
  const float* peb     = (const float*)d_in[9];
  const float* Wq      = (const float*)d_in[10];
  const float* Wk      = (const float*)d_in[11];
  const float* Wv      = (const float*)d_in[12];
  const float* Wo      = (const float*)d_in[13];
  const float* dw0     = (const float*)d_in[14];
  const float* db0     = (const float*)d_in[15];
  const float* dw1     = (const float*)d_in[16];
  const float* db1     = (const float*)d_in[17];
  const float* dw2     = (const float*)d_in[18];
  const float* db2     = (const float*)d_in[19];
  const float* dw3     = (const float*)d_in[20];
  const float* db3     = (const float*)d_in[21];
  const float* ln1g    = (const float*)d_in[22];
  const float* ln1b    = (const float*)d_in[23];
  const float* ln2g    = (const float*)d_in[24];
  const float* ln2b    = (const float*)d_in[25];
  const float* g1      = (const float*)d_in[26];
  const float* g2      = (const float*)d_in[27];
  const float* ffo     = (const float*)d_in[28];
  const float* lnfg    = (const float*)d_in[29];
  const float* lnfb    = (const float*)d_in[30];

  // workspace carving (floats)
  float* ws    = (float*)d_ws;
  float* dists = ws;                                    // B*N*N
  float* xbuf  = dists + (size_t)Bb * Nn * Nn;          // BN*D
  float* hbuf  = xbuf + (size_t)BN * Dd;                // BN*D
  float* sdbuf = hbuf + (size_t)BN * Dd;                // B*N*N
  float* obuf  = sdbuf + (size_t)Bb * Nn * Nn;          // BN*D
  float* ubuf  = obuf + (size_t)BN * Dd;                // BN*DFF
  _Float16* qb = (_Float16*)(ubuf + (size_t)BN * DFFf); // BN*D f16
  _Float16* kb = qb + (size_t)BN * Dd;
  _Float16* vb = kb + (size_t)BN * Dd;
  float* w1p   = (float*)(vb + (size_t)BN * Dd);        // L*64*64
  float* w2p   = w1p + Ll * HIDP * HIDP;
  float* w0p   = w2p + Ll * HIDP * HIDP;                // L*64 each below
  float* b0p   = w0p + Ll * HIDP;
  float* b1p   = b0p + Ll * HIDP;
  float* b2p   = b1p + Ll * HIDP;
  float* w3p   = b2p + Ll * HIDP;

  dim3 wv(32, 8);

  gt_prep<<<48, 256, 0, stream>>>(dw0, db0, dw1, db1, dw2, db2, dw3,
                                  w0p, b0p, w1p, b1p, w2p, b2p, w3p);
  gt_dists<<<Bb * Nn, 256, 0, stream>>>(R, amask, dists);
  gt_embed<<<Bb * Nn, 256, 0, stream>>>(charges, dists, pw0, pb0, pw1, pb1,
                                        pew, peb, aemb, xbuf);

  const int qkvBlk  = (BN / 16) * (Dd / 16) / 8;    // 1024 blocks (3 mats fused)
  const int gemmBlk = (BN / 16) * (Dd / 64) / 8;    // 256 blocks (16x64 strips)
  const int gateBlk = (BN / 16) * (DFFf / 32) / 8;  // 2048 blocks (16x32 strips)
  const int mlpBlk  = Bb * Nn * Nn / (16 * 8);      // 32768 blocks
  const int attBlk  = Bb * Hh * (Nn / 16) / 8;      // 256 blocks

  for (int l = 0; l < Ll; ++l) {
    gt_ln<<<BN / 8, wv, 0, stream>>>(xbuf, ln1g + l * Dd, ln1b + l * Dd, hbuf);
    gt_qkv<<<qkvBlk, wv, 0, stream>>>(hbuf, Wq + (size_t)l * Dd * Dd,
                                      Wk + (size_t)l * Dd * Dd,
                                      Wv + (size_t)l * Dd * Dd, qb, kb, vb);
    gt_distmlp<<<mlpBlk, wv, 0, stream>>>(dists, amask,
                                          w0p + l * HIDP, b0p + l * HIDP,
                                          w1p + l * HIDP * HIDP, b1p + l * HIDP,
                                          w2p + l * HIDP * HIDP, b2p + l * HIDP,
                                          w3p + l * HIDP, db3 + l, sdbuf);
    gt_attn<<<attBlk, wv, 0, stream>>>(qb, kb, vb, sdbuf, amask, obuf);
    gt_gemm<<<gemmBlk, wv, 0, stream>>>(obuf, Wo + (size_t)l * Dd * Dd, xbuf, xbuf,
                                        Dd, Dd);
    gt_ln<<<BN / 8, wv, 0, stream>>>(xbuf, ln2g + l * Dd, ln2b + l * Dd, hbuf);
    gt_gate<<<gateBlk, wv, 0, stream>>>(hbuf, g1 + (size_t)l * DFFf * Dd,
                                        g2 + (size_t)l * DFFf * Dd, ubuf);
    gt_gemm<<<gemmBlk, wv, 0, stream>>>(ubuf, ffo + (size_t)l * Dd * DFFf, xbuf, xbuf,
                                        Dd, DFFf);
  }
  gt_ln<<<BN / 8, wv, 0, stream>>>(xbuf, lnfg, lnfb, (float*)d_out);
}
